// TitansMAC_1486058684600
// MI455X (gfx1250) — matrix operand order repaired and checked by reading, not compiled, not run on hardware
//
#include <hip/hip_runtime.h>

typedef __attribute__((ext_vector_type(16))) _Float16 v16h;
typedef __attribute__((ext_vector_type(8)))  _Float16 v8h;
typedef __attribute__((ext_vector_type(8)))  float    v8f;

namespace {

constexpr int kS    = 2048;
constexpr int kD    = 256;
constexpr int kNH   = 8;
constexpr int kHD   = 32;   // head dim
constexpr int kP    = 8;    // persistent tokens
constexpr int kC    = 128;  // chunk
constexpr int kH    = 256;  // fast-memory hidden
constexpr int kCH   = 16;   // coeff hidden
constexpr int kL    = kP + 2 * kC;  // 264
constexpr int kLp   = 288;          // padded to multiple of 32
constexpr int kLT   = kLp / 16;     // 18 row tiles
constexpr int kNSEG = kS / kC;      // 16
constexpr int kBS   = 512;          // 16 wave32s
constexpr int kNW   = kBS / 32;

// ---------- per-batch workspace: f32 state region then f16 region ----------
// f32 offsets (floats)
constexpr size_t F_FW1 = 0;
constexpr size_t F_FB1 = F_FW1 + (size_t)kD * kH;
constexpr size_t F_FW2 = F_FB1 + kH;
constexpr size_t F_FB2 = F_FW2 + (size_t)kH * kD;
constexpr size_t F_MW1 = F_FB2 + kD;
constexpr size_t F_MB1 = F_MW1 + (size_t)kD * kH;
constexpr size_t F_MW2 = F_MB1 + kH;
constexpr size_t F_MB2 = F_MW2 + (size_t)kH * kD;
constexpr size_t F_GW1 = F_MB2 + kD;
constexpr size_t F_GB1 = F_GW1 + (size_t)kD * kH;
constexpr size_t F_GW2 = F_GB1 + kH;
constexpr size_t F_GB2 = F_GW2 + (size_t)kH * kD;
constexpr size_t WS_F32 = F_GB2 + kD;           // 394752 floats
constexpr size_t F32_BYTES = WS_F32 * 4;        // multiple of 64

// f16 offsets (halves) -- every buffer size is a multiple of 16 halves (32B alignment)
constexpr size_t H_XF  = 0;                               // 128x256 staged x segment
constexpr size_t H_Q   = H_XF  + (size_t)kC * kD;         // 128x256
constexpr size_t H_AIN = H_Q   + (size_t)kC * kD;         // 288x256
constexpr size_t H_QKV = H_AIN + (size_t)kLp * kD;        // 288x768
constexpr size_t H_AO  = H_QKV + (size_t)kLp * 3 * kD;    // 288x256 (attn out / mem_out)
constexpr size_t H_Y   = H_AO  + (size_t)kLp * kD;
constexpr size_t H_K   = H_Y   + (size_t)kLp * kD;
constexpr size_t H_V   = H_K   + (size_t)kLp * kD;
constexpr size_t H_ZK  = H_V   + (size_t)kLp * kD;        // pre-activations (reused)
constexpr size_t H_HK  = H_ZK  + (size_t)kLp * kH;        // hidden (reused as dh/dz)
constexpr size_t H_PRD = H_HK  + (size_t)kLp * kH;        // pred / ds = pred - v
constexpr size_t H_FW1P  = H_PRD  + (size_t)kLp * kD;     // packed fast W1   (B-frag native)
constexpr size_t H_FW2P  = H_FW1P + (size_t)kH * kD;      // packed fast W2
constexpr size_t H_FW2TP = H_FW2P + (size_t)kH * kD;      // packed fast W2^T
constexpr size_t H_WQP   = H_FW2TP+ (size_t)kH * kD;      // packed W_Q
constexpr size_t H_WKP   = H_WQP  + (size_t)kD * kD;      // packed W_K
constexpr size_t H_WVP   = H_WKP  + (size_t)kD * kD;      // packed W_V
constexpr size_t H_IPWP  = H_WVP  + (size_t)kD * kD;      // packed in_proj^T (256x768)
constexpr size_t H_OPWP  = H_IPWP + (size_t)kD * 3 * kD;  // packed out_proj^T
constexpr size_t WS_H16  = H_OPWP + (size_t)kD * kD;
constexpr size_t WS_PER_B_BYTES = F32_BYTES + WS_H16 * 2;

__device__ inline float sigm(float x)  { return 1.0f / (1.0f + __expf(-x)); }
__device__ inline float silu(float x)  { return x * sigm(x); }
__device__ inline float dsilu(float x) { float s = sigm(x); return s * (1.0f + x * (1.0f - s)); }

// ---- pack a (K x N) f32/f16 matrix (optionally from its (N x K) transpose) into
// ---- WMMA B-fragment-native order: dst[((nt*KT + kt)*32 + lane)*16 + e],
// ---- element = B[kt*32 + 16*(e>>3) + 8g + (e&7)][nt*16 + (lane&15)]
template<typename T>
__device__ void pack_b(const T* __restrict__ W, int ldw, bool trans,
                       _Float16* __restrict__ dst, int KT, int NT) {
  const int tot = KT * NT * 512;
  for (int i = threadIdx.x; i < tot; i += kBS) {
    int e = i & 15, lane = (i >> 4) & 31, tile = i >> 9;
    int kt = tile % KT, nt = tile / KT;
    int gg = (lane >> 4) & 1;
    int k = (kt << 5) + ((e >> 3) << 4) + (gg << 3) + (e & 7);
    int n = (nt << 4) + (lane & 15);
    float v = trans ? (float)W[(size_t)n * ldw + k] : (float)W[(size_t)k * ldw + n];
    dst[i] = (_Float16)v;
  }
}

// ---- fast GEMM: A = f16 row-major (b128 loads), B = packed fragment-native ----
// Out (f16) = (A @ B + bias) * oscale ; rows >= Mvalid zeroed
__device__ void gemm_fast(const _Float16* __restrict__ A, int lda,
                          const _Float16* __restrict__ Bp,
                          const float* __restrict__ bias,
                          _Float16* __restrict__ Out, int ldo,
                          int Mtiles, int Ntiles, int K, int Mvalid, float oscale) {
  const int tid  = threadIdx.x;
  const int wave = tid >> 5;
  const int lane = tid & 31;
  const int g    = (lane >> 4) & 1;
  const int KT   = K >> 5;
  const int nj   = Mtiles * Ntiles;
  for (int job = wave; job < nj; job += kNW) {
    const int m0 = (job / Ntiles) << 4;
    const int nt = job % Ntiles;
    const int n0 = nt << 4;
    float bv = bias ? bias[n0 + (lane & 15)] : 0.0f;
    v8f acc;
#pragma unroll
    for (int j = 0; j < 8; ++j) acc[j] = bv;
    const _Float16* arow = A + (size_t)(m0 + (lane & 15)) * lda + (g << 3);
    const _Float16* bp   = Bp + ((size_t)nt * KT * 32 + lane) * 16;
    for (int kt = 0; kt < KT; ++kt) {
      v8h lo = *(const v8h*)(arow + (kt << 5));
      v8h hi = *(const v8h*)(arow + (kt << 5) + 16);
      v16h a;
#pragma unroll
      for (int e = 0; e < 8; ++e) { a[e] = lo[e]; a[8 + e] = hi[e]; }
      v16h bf = *(const v16h*)(bp + (size_t)kt * 512);
      acc = __builtin_amdgcn_wmma_f32_16x16x32_f16(false, a, false, bf, (short)0, acc, false, false);
    }
#pragma unroll
    for (int j = 0; j < 8; ++j) {
      int r = m0 + j + (g << 3);
      Out[(size_t)r * ldo + n0 + (lane & 15)] =
          (r < Mvalid) ? (_Float16)(acc[j] * oscale) : (_Float16)0.0f;
    }
  }
}

// ---- gradient GEMM: Out(f32) = (A^T @ B) * oscale ; A,B f16 row-major ----
__device__ void gemm_grad(const _Float16* __restrict__ A, int lda,
                          const _Float16* __restrict__ B, int ldb,
                          float* __restrict__ Out, int ldo,
                          int Mtiles, int Ntiles, int K, float oscale) {
  const int tid  = threadIdx.x;
  const int wave = tid >> 5;
  const int lane = tid & 31;
  const int g    = (lane >> 4) & 1;
  const int nj   = Mtiles * Ntiles;
  for (int job = wave; job < nj; job += kNW) {
    const int m0 = (job / Ntiles) << 4;
    const int n0 = (job % Ntiles) << 4;
    v8f acc;
#pragma unroll
    for (int j = 0; j < 8; ++j) acc[j] = 0.0f;
    for (int k0 = 0; k0 < K; k0 += 32) {
      v16h a, bf;
#pragma unroll
      for (int e = 0; e < 16; ++e) {
        int k = k0 + ((e >> 3) << 4) + (g << 3) + (e & 7);
        a[e] = A[(size_t)k * lda + m0 + (lane & 15)];
      }
#pragma unroll
      for (int e = 0; e < 16; ++e) {
        int k = k0 + ((e >> 3) << 4) + (g << 3) + (e & 7);
        bf[e] = B[(size_t)k * ldb + n0 + (lane & 15)];
      }
      acc = __builtin_amdgcn_wmma_f32_16x16x32_f16(false, a, false, bf, (short)0, acc, false, false);
    }
#pragma unroll
    for (int j = 0; j < 8; ++j) {
      int r = m0 + j + (g << 3);
      Out[(size_t)r * ldo + n0 + (lane & 15)] = acc[j] * oscale;
    }
  }
}

__global__ __launch_bounds__(kBS)
void titans_mac_kernel(const float* __restrict__ x,
                       const float* __restrict__ pers,
                       const float* __restrict__ WQ,
                       const float* __restrict__ ipw,
                       const float* __restrict__ ipb,
                       const float* __restrict__ opw,
                       const float* __restrict__ opb,
                       const float* __restrict__ mW1_in,
                       const float* __restrict__ mb1_in,
                       const float* __restrict__ mW2_in,
                       const float* __restrict__ mb2_in,
                       const float* __restrict__ WK,
                       const float* __restrict__ WV,
                       const float* __restrict__ cW1,
                       const float* __restrict__ cb1,
                       const float* __restrict__ cW2,
                       const float* __restrict__ cb2,
                       float* __restrict__ out,
                       char* __restrict__ ws) {
  const int b    = blockIdx.x;
  const int tid  = threadIdx.x;
  const int wave = tid >> 5;
  const int lane = tid & 31;
  const int g    = (lane >> 4) & 1;

  char* wbase = ws + (size_t)b * WS_PER_B_BYTES;
  float*    fr = (float*)wbase;
  _Float16* hr = (_Float16*)(wbase + F32_BYTES);

  float* fw1 = fr + F_FW1;  float* fb1  = fr + F_FB1;
  float* fw2 = fr + F_FW2;  float* fb2  = fr + F_FB2;
  float* mo1 = fr + F_MW1;  float* mob1 = fr + F_MB1;
  float* mo2 = fr + F_MW2;  float* mob2 = fr + F_MB2;
  float* gw1 = fr + F_GW1;  float* gb1  = fr + F_GB1;
  float* gw2 = fr + F_GW2;  float* gb2  = fr + F_GB2;

  _Float16* xf   = hr + H_XF;
  _Float16* qb   = hr + H_Q;
  _Float16* ainb = hr + H_AIN;
  _Float16* qkvb = hr + H_QKV;
  _Float16* aob  = hr + H_AO;
  _Float16* yb   = hr + H_Y;
  _Float16* kb   = hr + H_K;
  _Float16* vb   = hr + H_V;
  _Float16* zkb  = hr + H_ZK;
  _Float16* hkb  = hr + H_HK;
  _Float16* prdb = hr + H_PRD;
  _Float16* fw1p = hr + H_FW1P;
  _Float16* fw2p = hr + H_FW2P;
  _Float16* fw2tp= hr + H_FW2TP;
  _Float16* wqp  = hr + H_WQP;
  _Float16* wkp  = hr + H_WKP;
  _Float16* wvp  = hr + H_WVP;
  _Float16* ipwp = hr + H_IPWP;
  _Float16* opwp = hr + H_OPWP;

  __shared__ _Float16 pls[kNW][16][32];  // per-wave P-tile (accum -> A-frag relayout)
  __shared__ float ymean[kD];
  __shared__ float chs[kCH];
  __shared__ float coefv[4];

  // ---- init recurrent state: fast = broadcast(mW*), mom = 0 ----
  for (int i = tid; i < kD * kH; i += kBS) {
    fw1[i] = mW1_in[i]; mo1[i] = 0.0f;
    fw2[i] = mW2_in[i]; mo2[i] = 0.0f;
  }
  for (int i = tid; i < kD; i += kBS) {
    fb1[i] = mb1_in[i]; mob1[i] = 0.0f;
    fb2[i] = mb2_in[i]; mob2[i] = 0.0f;
  }
  __syncthreads();
  // ---- one-time packs: static weights + initial fast weights ----
  pack_b(WQ,  kD, false, wqp,  kD / 32, kD / 16);
  pack_b(WK,  kD, false, wkp,  kD / 32, kD / 16);
  pack_b(WV,  kD, false, wvp,  kD / 32, kD / 16);
  pack_b(ipw, kD, true,  ipwp, kD / 32, 3 * kD / 16);
  pack_b(opw, kD, true,  opwp, kD / 32, kD / 16);
  pack_b(fw1, kH, false, fw1p, kD / 32, kH / 16);
  pack_b(fw2, kD, false, fw2p, kH / 32, kD / 16);
  pack_b(fw2, kD, true,  fw2tp, kD / 32, kH / 16);
  __syncthreads();

  const float invTD  = 1.0f / ((float)kL * (float)kD);
  const float rscale = 0.17677669529663687f;  // 1/sqrt(32)

  for (int seg = 0; seg < kNSEG; ++seg) {
    const float* xseg = x + ((size_t)b * kS + (size_t)seg * kC) * kD;

    if (seg + 1 < kNSEG) {
      const float* xn = xseg + (size_t)kC * kD;
      for (int i = tid * 32; i < kC * kD; i += kBS * 32)
        __builtin_prefetch(xn + i, 0, 1);
    }

    // stage x segment to f16
    for (int i = tid; i < kC * kD; i += kBS) xf[i] = (_Float16)xseg[i];
    __syncthreads();
    // q = seg @ W_Q
    gemm_fast(xf, kD, wqp, nullptr, qb, kD, kC / 16, kD / 16, kD, kC, 1.0f);
    __syncthreads();
    // zq = q @ W1 + b1 (fast, pre-update)
    gemm_fast(qb, kD, fw1p, fb1, zkb, kH, kC / 16, kH / 16, kD, kC, 1.0f);
    __syncthreads();
    // attn_in = [persistent ; silu(zq) ; seg ; 0-pad]
    for (int i = tid; i < kLp * kD; i += kBS) {
      int t = i >> 8, d = i & (kD - 1);
      _Float16 v;
      if      (t < kP)      v = (_Float16)pers[(size_t)t * kD + d];
      else if (t < kP + kC) v = (_Float16)silu((float)zkb[(size_t)(t - kP) * kH + d]);
      else if (t < kL)      v = xf[(size_t)(t - kP - kC) * kD + d];
      else                  v = (_Float16)0.0f;
      ainb[i] = v;
    }
    __syncthreads();
    // qkv = attn_in @ in_proj_w^T + in_proj_b
    gemm_fast(ainb, kD, ipwp, ipb, qkvb, 3 * kD, kLT, 3 * kD / 16, kD, kL, 1.0f);
    __syncthreads();

    // ---- causal flash attention per (head, 16-row query tile) ----
    {
      const int jobs = kNH * kLT;  // 144
      for (int job = wave; job < jobs; job += kNW) {
        const int hh = job / kLT;
        const int m0 = (job % kLT) * 16;
        const int hq = hh * kHD;
        // Q fragment: two contiguous b128 loads
        v16h aq;
        {
          const _Float16* qrow = qkvb + (size_t)(m0 + (lane & 15)) * (3 * kD) + hq + (g << 4);
          v8h lo = *(const v8h*)qrow;
          v8h hi = *(const v8h*)(qrow + 8);
#pragma unroll
          for (int e = 0; e < 8; ++e) { aq[e] = lo[e]; aq[8 + e] = hi[e]; }
        }
        float mrow[8], lrow[8];
        v8f o0, o1;
#pragma unroll
        for (int j = 0; j < 8; ++j) { mrow[j] = -3.0e38f; lrow[j] = 0.0f; o0[j] = 0.0f; o1[j] = 0.0f; }

        for (int kb0 = 0; kb0 < kLp; kb0 += 32) {
          // K^T fragments: per-lane contiguous 32B (head-dim is K-dim)
          v16h bk0 = *(const v16h*)(qkvb + (size_t)(kb0 + (lane & 15)) * (3 * kD) + kD + hq + (g << 4));
          v16h bk1 = *(const v16h*)(qkvb + (size_t)(kb0 + 16 + (lane & 15)) * (3 * kD) + kD + hq + (g << 4));
          v8f z0, z1;
#pragma unroll
          for (int j = 0; j < 8; ++j) { z0[j] = 0.0f; z1[j] = 0.0f; }
          v8f s0 = __builtin_amdgcn_wmma_f32_16x16x32_f16(false, aq, false, bk0, (short)0, z0, false, false);
          v8f s1 = __builtin_amdgcn_wmma_f32_16x16x32_f16(false, aq, false, bk1, (short)0, z1, false, false);

          const int cb0 = kb0 + (lane & 15);
#pragma unroll
          for (int j = 0; j < 8; ++j) {
            const int r = m0 + j + (g << 3);
            float v0 = s0[j] * rscale; if (cb0 > r)      v0 = -1.0e30f;  // causal (+pad) mask
            float v1 = s1[j] * rscale; if (cb0 + 16 > r) v1 = -1.0e30f;
            float bm = fmaxf(v0, v1);
#pragma unroll
            for (int mm = 1; mm < 16; mm <<= 1) bm = fmaxf(bm, __shfl_xor(bm, mm, 32));
            float mnew = fmaxf(mrow[j], bm);
            float sc   = __expf(mrow[j] - mnew);
            float p0   = __expf(v0 - mnew);
            float p1   = __expf(v1 - mnew);
            float rs   = p0 + p1;
#pragma unroll
            for (int mm = 1; mm < 16; mm <<= 1) rs += __shfl_xor(rs, mm, 32);
            lrow[j] = lrow[j] * sc + rs;
            mrow[j] = mnew;
            o0[j] *= sc; o1[j] *= sc;
            pls[wave][j + (g << 3)][lane & 15]        = (_Float16)p0;
            pls[wave][j + (g << 3)][16 + (lane & 15)] = (_Float16)p1;
          }
          // P (accum layout) -> A fragment via per-wave LDS tile
          v16h ap;
#pragma unroll
          for (int e = 0; e < 16; ++e) {
            int k = (g << 4) + e;
            ap[e] = pls[wave][lane & 15][k];
          }
          // V fragments (K-dim = 32 keys, strided)
          v16h bv0, bv1;
#pragma unroll
          for (int e = 0; e < 16; ++e) {
            int kk = kb0 + (g << 4) + e;
            bv0[e] = qkvb[(size_t)kk * (3 * kD) + 2 * kD + hq + (lane & 15)];
            bv1[e] = qkvb[(size_t)kk * (3 * kD) + 2 * kD + hq + 16 + (lane & 15)];
          }
          o0 = __builtin_amdgcn_wmma_f32_16x16x32_f16(false, ap, false, bv0, (short)0, o0, false, false);
          o1 = __builtin_amdgcn_wmma_f32_16x16x32_f16(false, ap, false, bv1, (short)0, o1, false, false);
        }
#pragma unroll
        for (int j = 0; j < 8; ++j) {
          const int r = m0 + j + (g << 3);
          const float inv = 1.0f / lrow[j];
          aob[(size_t)r * kD + hq + (lane & 15)]      = (_Float16)(o0[j] * inv);
          aob[(size_t)r * kD + hq + 16 + (lane & 15)] = (_Float16)(o1[j] * inv);
        }
      }
    }
    __syncthreads();

    // y = ao @ out_proj_w^T + out_proj_b
    gemm_fast(aob, kD, opwp, opb, yb, kD, kLT, kD / 16, kD, kL, 1.0f);
    __syncthreads();
    // k = y @ W_K ; v = y @ W_V
    gemm_fast(yb, kD, wkp, nullptr, kb, kD, kLT, kD / 16, kD, kL, 1.0f);
    gemm_fast(yb, kD, wvp, nullptr, vb, kD, kLT, kD / 16, kD, kL, 1.0f);
    __syncthreads();
    // zk = k @ W1 + b1
    gemm_fast(kb, kD, fw1p, fb1, zkb, kH, kLT, kH / 16, kD, kL, 1.0f);
    __syncthreads();
    // hk = silu(zk)
    for (int i = tid; i < kLp * kH; i += kBS) hkb[i] = (_Float16)silu((float)zkb[i]);
    __syncthreads();
    // pred = hk @ W2 + b2
    gemm_fast(hkb, kH, fw2p, fb2, prdb, kD, kLT, kD / 16, kH, kL, 1.0f);
    __syncthreads();
    // ds = pred - v   (unscaled; 1/(L*D) folded into gradient stores to avoid f16 underflow)
    for (int i = tid; i < kLp * kD; i += kBS)
      prdb[i] = (_Float16)((float)prdb[i] - (float)vb[i]);
    __syncthreads();
    // gW2 = (hk^T @ ds) * invTD ; gb2 = colsum(ds) * invTD
    gemm_grad(hkb, kH, prdb, kD, gw2, kD, kH / 16, kD / 16, kLp, invTD);
    for (int n = tid; n < kD; n += kBS) {
      float s = 0.0f;
      for (int t = 0; t < kL; ++t) s += (float)prdb[(size_t)t * kD + n];
      gb2[n] = s * invTD;
    }
    __syncthreads();
    // dh = ds @ W2^T
    gemm_fast(prdb, kD, fw2tp, nullptr, hkb, kH, kLT, kH / 16, kD, kL, 1.0f);
    __syncthreads();
    // dz = dh * silu'(zk)
    for (int i = tid; i < kLp * kH; i += kBS)
      hkb[i] = (_Float16)((float)hkb[i] * dsilu((float)zkb[i]));
    __syncthreads();
    // gW1 = (k^T @ dz) * invTD ; gb1 = colsum(dz) * invTD
    gemm_grad(kb, kD, hkb, kH, gw1, kH, kD / 16, kH / 16, kLp, invTD);
    for (int n = tid; n < kH; n += kBS) {
      float s = 0.0f;
      for (int t = 0; t < kL; ++t) s += (float)hkb[(size_t)t * kH + n];
      gb1[n] = s * invTD;
    }
    __syncthreads();
    // gating coefficients from mean(y)
    for (int d = tid; d < kD; d += kBS) {
      float s = 0.0f;
      for (int t = 0; t < kL; ++t) s += (float)yb[(size_t)t * kD + d];
      ymean[d] = s * (1.0f / (float)kL);
    }
    __syncthreads();
    if (tid < kCH) {
      float s = cb1[tid];
      for (int d = 0; d < kD; ++d) s += ymean[d] * cW1[(size_t)d * kCH + tid];
      chs[tid] = silu(s);
    }
    __syncthreads();
    if (tid < 3) {
      float s = cb2[tid];
      for (int i = 0; i < kCH; ++i) s += chs[i] * cW2[(size_t)i * 3 + tid];
      coefv[tid] = sigm(s);
    }
    __syncthreads();
    const float alpha = coefv[0], eta = coefv[1], theta = coefv[2];
    const float keep = 1.0f - alpha;
    // momentum + fast-weight update (f32 master state)
    for (int i = tid; i < kD * kH; i += kBS) {
      float m1 = eta * mo1[i] - theta * gw1[i]; mo1[i] = m1; fw1[i] = keep * fw1[i] + m1;
      float m2 = eta * mo2[i] - theta * gw2[i]; mo2[i] = m2; fw2[i] = keep * fw2[i] + m2;
    }
    for (int i = tid; i < kD; i += kBS) {
      float m1 = eta * mob1[i] - theta * gb1[i]; mob1[i] = m1; fb1[i] = keep * fb1[i] + m1;
      float m2 = eta * mob2[i] - theta * gb2[i]; mob2[i] = m2; fb2[i] = keep * fb2[i] + m2;
    }
    __syncthreads();
    // repack updated fast weights into fragment-native f16
    pack_b(fw1, kH, false, fw1p, kD / 32, kH / 16);
    pack_b(fw2, kD, false, fw2p, kH / 32, kD / 16);
    pack_b(fw2, kD, true,  fw2tp, kD / 32, kH / 16);
    __syncthreads();
    // mem_out = mem_apply(fast_new, y)
    gemm_fast(yb, kD, fw1p, fb1, zkb, kH, kLT, kH / 16, kD, kL, 1.0f);
    __syncthreads();
    for (int i = tid; i < kLp * kH; i += kBS) hkb[i] = (_Float16)silu((float)zkb[i]);
    __syncthreads();
    gemm_fast(hkb, kH, fw2p, fb2, aob, kD, kLT, kD / 16, kH, kL, 1.0f);
    __syncthreads();
    // o = y * mem_out, keep last C rows
    for (int i = tid; i < kC * kD; i += kBS) {
      int tt = i >> 8, d = i & (kD - 1);
      int t = kP + kC + tt;
      out[((size_t)b * kS + (size_t)seg * kC + tt) * kD + d] =
          (float)yb[(size_t)t * kD + d] * (float)aob[(size_t)t * kD + d];
    }
    __syncthreads();
  }
}

}  // namespace

extern "C" void kernel_launch(void* const* d_in, const int* in_sizes, int n_in,
                              void* d_out, int out_size, void* d_ws, size_t ws_size,
                              hipStream_t stream) {
  (void)in_sizes; (void)n_in; (void)out_size; (void)ws_size;
  const float* x    = (const float*)d_in[0];
  const float* pers = (const float*)d_in[1];
  const float* WQ   = (const float*)d_in[2];
  const float* ipw  = (const float*)d_in[3];
  const float* ipb  = (const float*)d_in[4];
  const float* opw  = (const float*)d_in[5];
  const float* opb  = (const float*)d_in[6];
  const float* mW1  = (const float*)d_in[7];
  const float* mb1  = (const float*)d_in[8];
  const float* mW2  = (const float*)d_in[9];
  const float* mb2  = (const float*)d_in[10];
  const float* WK   = (const float*)d_in[11];
  const float* WV   = (const float*)d_in[12];
  const float* cW1  = (const float*)d_in[13];
  const float* cb1  = (const float*)d_in[14];
  const float* cW2  = (const float*)d_in[15];
  const float* cb2  = (const float*)d_in[16];

  titans_mac_kernel<<<dim3(16), dim3(kBS), 0, stream>>>(
      x, pers, WQ, ipw, ipb, opw, opb, mW1, mb1, mW2, mb2, WK, WV,
      cW1, cb1, cW2, cb2, (float*)d_out, (char*)d_ws);
}
